// CausalSelfAttention_39926015983894
// MI455X (gfx1250) — matrix operand order repaired and checked by reading, not compiled, not run on hardware
//
#include <hip/hip_runtime.h>

#define EMB   1024
#define NH    16
#define HD    64
#define BB    4
#define TSEQ  2048
#define NTOK  (BB * TSEQ)       // 8192 tokens
#define KDIM  1024

typedef __attribute__((ext_vector_type(8)))  __bf16 v8bf;
typedef __attribute__((ext_vector_type(16))) __bf16 v16bf;
typedef __attribute__((ext_vector_type(8)))  float  v8f;
typedef int b128_t __attribute__((vector_size(16)));   // matches builtin's int4

// ---------------------------------------------------------------------------
// CDNA5 async global->LDS copy (ASYNCcnt-tracked, bypasses VGPRs)
// ---------------------------------------------------------------------------
#if __has_builtin(__builtin_amdgcn_global_load_async_to_lds_b128)
#define HAVE_ASYNC_LDS 1
#else
#define HAVE_ASYNC_LDS 0
#endif

__device__ __forceinline__ void cp16(const __bf16* g, __bf16* l) {
#if HAVE_ASYNC_LDS
  __builtin_amdgcn_global_load_async_to_lds_b128(
      (__attribute__((address_space(1))) b128_t*)g,
      (__attribute__((address_space(3))) b128_t*)l, 0, 0);
#else
  *(v8bf*)l = *(const v8bf*)g;
#endif
}

__device__ __forceinline__ void wait_async0() {
#if HAVE_ASYNC_LDS
#if __has_builtin(__builtin_amdgcn_s_wait_asynccnt)
  __builtin_amdgcn_s_wait_asynccnt(0);
#else
  asm volatile("s_wait_asynccnt 0x0" ::: "memory");
#endif
#endif
}

// ---------------------------------------------------------------------------
// helpers
// ---------------------------------------------------------------------------
__device__ __forceinline__ v16bf ldpair(const __bf16* p0, const __bf16* p1) {
  v8bf lo = *(const v8bf*)p0;
  v8bf hi = *(const v8bf*)p1;
  return __builtin_shufflevector(lo, hi, 0, 1, 2, 3, 4, 5, 6, 7,
                                 8, 9, 10, 11, 12, 13, 14, 15);
}

__device__ __forceinline__ v8f wmma_bf16(v16bf a, v16bf b, v8f c) {
  // D = A(16x32 bf16) * B(32x16 bf16) + C(16x16 f32)
  return __builtin_amdgcn_wmma_f32_16x16x32_bf16(false, a, false, b,
                                                 (short)0, c, false, false);
}

// ---------------------------------------------------------------------------
// absmean scale (deterministic two-pass reduction) + ternary quantization
// ---------------------------------------------------------------------------
__global__ __launch_bounds__(256) void absum_partial(const float* __restrict__ w,
                                                     int n, float* __restrict__ part) {
  __shared__ float sm[256];
  float s = 0.0f;
  for (int i = blockIdx.x * 256 + threadIdx.x; i < n; i += 256 * gridDim.x)
    s += fabsf(w[i]);
  sm[threadIdx.x] = s;
  __syncthreads();
  for (int st = 128; st > 0; st >>= 1) {
    if (threadIdx.x < st) sm[threadIdx.x] += sm[threadIdx.x + st];
    __syncthreads();
  }
  if (threadIdx.x == 0) part[blockIdx.x] = sm[0];
}

__global__ __launch_bounds__(256) void absum_final(const float* __restrict__ part,
                                                   int n, float* __restrict__ scale) {
  __shared__ float sm[256];
  sm[threadIdx.x] = part[threadIdx.x];
  __syncthreads();
  for (int st = 128; st > 0; st >>= 1) {
    if (threadIdx.x < st) sm[threadIdx.x] += sm[threadIdx.x + st];
    __syncthreads();
  }
  if (threadIdx.x == 0) *scale = fmaxf(sm[0] / (float)n, 1e-8f);
}

__global__ __launch_bounds__(256) void quantize_ternary(const float* __restrict__ w,
                                                        const float* __restrict__ scale,
                                                        __bf16* __restrict__ out, int n) {
  int i = blockIdx.x * 256 + threadIdx.x;
  if (i >= n) return;
  float inv = 1.0f / *scale;
  float t = rintf(w[i] * inv);           // round-half-even, matches jnp.round
  t = fminf(1.0f, fmaxf(-1.0f, t));
  out[i] = (__bf16)t;
}

__global__ __launch_bounds__(256) void cvt_f32_bf16(const float* __restrict__ x,
                                                    __bf16* __restrict__ y, int n) {
  int i = blockIdx.x * 256 + threadIdx.x;
  if (i < n) y[i] = (__bf16)x[i];
}

// ---------------------------------------------------------------------------
// GEMM core: C[128x128] tile of  A[M x K] * W[N x K]^T  (both bf16 row-major)
// 256 threads = 8 waves; wave grid 2(M,64 rows) x 4(N,32 cols);
// per wave: 4x2 WMMA accumulators, K-step 32, DOUBLE-BUFFERED async staging.
// ---------------------------------------------------------------------------
#define GK   32
#define LDP  40            // padded LDS row stride (elements) to spread banks
#define TILE_ELEMS (128 * LDP)

__device__ __forceinline__ void stage_tiles(const __bf16* __restrict__ A,
                                            const __bf16* __restrict__ W,
                                            int K, size_t arow0, size_t wrow0,
                                            int kk, __bf16* lA, __bf16* lW) {
  const int tid = threadIdx.x;
  for (int i = tid; i < (128 * GK) / 8; i += 256) {   // 512 chunks -> 2/thread
    int r = i >> 2, c = (i & 3) * 8;
    cp16(&A[(arow0 + r) * (size_t)K + kk + c], &lA[r * LDP + c]);
    cp16(&W[(wrow0 + r) * (size_t)K + kk + c], &lW[r * LDP + c]);
  }
}

__device__ __forceinline__ void gemm_tile_core(const __bf16* __restrict__ A,
                                               const __bf16* __restrict__ W,
                                               int K, size_t arow0, size_t wrow0,
                                               __bf16* lA, __bf16* lW,   // 2 buffers each
                                               v8f acc[4][2]) {
  const int tid = threadIdx.x;
  const int wave = tid >> 5, lane = tid & 31;
  const int hh = lane >> 4, ln = lane & 15;
  const int wm = wave >> 2, wn = wave & 3;
  const int nsteps = K / GK;

  // prologue: fill buffer 0
  stage_tiles(A, W, K, arow0, wrow0, 0, lA, lW);
  wait_async0();
  __syncthreads();

  for (int s = 0; s < nsteps; s++) {
    const int cur = s & 1;
    __bf16* cA = lA + cur * TILE_ELEMS;
    __bf16* cW = lW + cur * TILE_ELEMS;

    // prefetch next tile into the other buffer while computing this one
    if (s + 1 < nsteps)
      stage_tiles(A, W, K, arow0, wrow0, (s + 1) * GK,
                  lA + (1 - cur) * TILE_ELEMS, lW + (1 - cur) * TILE_ELEMS);

    v16bf af[4];
#pragma unroll
    for (int mt = 0; mt < 4; mt++) {
      const __bf16* p = &cA[(wm * 64 + mt * 16 + ln) * LDP + hh * 8];
      af[mt] = ldpair(p, p + 16);          // A-frag: k0+half*8 / k0+16+half*8
    }
#pragma unroll
    for (int nt = 0; nt < 2; nt++) {
      const __bf16* p = &cW[(wn * 32 + nt * 16 + ln) * LDP + hh * 8];
      v16bf bfrag = ldpair(p, p + 16);     // B-frag: k0+half*8 / k0+16+half*8
#pragma unroll
      for (int mt = 0; mt < 4; mt++)
        acc[mt][nt] = wmma_bf16(af[mt], bfrag, acc[mt][nt]);
    }

    wait_async0();       // our prefetch done; barrier covers all waves
    __syncthreads();
  }
}

// QKV projection: X[8192x1024] * Wq[3072x1024]^T; scatter bf16 into
// q/k/v buffers laid out [B, H, T, D]; fold 1/sqrt(HD)=0.125 into Q.
__global__ __launch_bounds__(256) void gemm_qkv(const __bf16* __restrict__ X,
                                                const __bf16* __restrict__ Wq,
                                                __bf16* __restrict__ qb,
                                                __bf16* __restrict__ kb,
                                                __bf16* __restrict__ vb) {
  __shared__ __align__(64) __bf16 lA[2 * TILE_ELEMS];
  __shared__ __align__(64) __bf16 lW[2 * TILE_ELEMS];
  v8f acc[4][2] = {};
  size_t arow0 = (size_t)blockIdx.x * 128;
  size_t wrow0 = (size_t)blockIdx.y * 128;
  gemm_tile_core(X, Wq, KDIM, arow0, wrow0, lA, lW, acc);

  const int tid = threadIdx.x;
  const int wave = tid >> 5, lane = tid & 31;
  const int hh = lane >> 4, ln = lane & 15;
  const int wm = wave >> 2, wn = wave & 3;

#pragma unroll
  for (int nt = 0; nt < 2; nt++) {
    int col = (int)wrow0 + wn * 32 + nt * 16 + ln;  // 0..3071
    int part = col >> 10;                           // 0=q 1=k 2=v
    int cc = col & 1023;
    int h = cc >> 6, d = cc & 63;
    __bf16* dst = (part == 0) ? qb : ((part == 1) ? kb : vb);
    float sc = (part == 0) ? 0.125f : 1.0f;
#pragma unroll
    for (int mt = 0; mt < 4; mt++) {
#pragma unroll
      for (int j = 0; j < 8; j++) {
        int row = (int)arow0 + wm * 64 + mt * 16 + j + 8 * hh;  // token 0..8191
        int bidx = row >> 11, t = row & 2047;
        dst[((((size_t)bidx * NH + h) * TSEQ) + t) * HD + d] =
            (__bf16)(acc[mt][nt][j] * sc);
      }
    }
  }
}

// Output projection: Attn[8192x1024] * Wo[1024x1024]^T -> f32 out
__global__ __launch_bounds__(256) void gemm_out(const __bf16* __restrict__ X,
                                                const __bf16* __restrict__ Wo,
                                                float* __restrict__ out) {
  __shared__ __align__(64) __bf16 lA[2 * TILE_ELEMS];
  __shared__ __align__(64) __bf16 lW[2 * TILE_ELEMS];
  v8f acc[4][2] = {};
  size_t arow0 = (size_t)blockIdx.x * 128;
  size_t wrow0 = (size_t)blockIdx.y * 128;
  gemm_tile_core(X, Wo, KDIM, arow0, wrow0, lA, lW, acc);

  const int tid = threadIdx.x;
  const int wave = tid >> 5, lane = tid & 31;
  const int hh = lane >> 4, ln = lane & 15;
  const int wm = wave >> 2, wn = wave & 3;

#pragma unroll
  for (int nt = 0; nt < 2; nt++) {
    int col = (int)wrow0 + wn * 32 + nt * 16 + ln;
#pragma unroll
    for (int mt = 0; mt < 4; mt++) {
#pragma unroll
      for (int j = 0; j < 8; j++) {
        int row = (int)arow0 + wm * 64 + mt * 16 + j + 8 * hh;
        out[(size_t)row * EMB + col] = acc[mt][nt][j];
      }
    }
  }
}

// ---------------------------------------------------------------------------
// Flash attention. Block = 128 threads (4 waves) = 64 query rows of one (b,h).
// Each wave owns 16 rows: O accum 16x64 f32, online softmax stats per row.
// K block 64x64 row-major in LDS; V block stored TRANSPOSED (d-major) in LDS.
// ---------------------------------------------------------------------------
#define BQ 64
#define BK 64

__global__ __launch_bounds__(128) void flash_attn(const __bf16* __restrict__ qb,
                                                  const __bf16* __restrict__ kb,
                                                  const __bf16* __restrict__ vb,
                                                  __bf16* __restrict__ attn_out) {
  __shared__ __align__(64) __bf16 lk[BK * HD];        // K rows  [n][d]
  __shared__ __align__(64) __bf16 lvt[HD * BK];       // V^T     [d][n]
  __shared__ __align__(64) __bf16 lp[4][16 * BK];     // per-wave P tile

  const int bh = blockIdx.x;                 // b*NH + h
  const int b = bh / NH, h = bh % NH;
  const int qblk = blockIdx.y * BQ;
  const int tid = threadIdx.x;
  const int wave = tid >> 5, lane = tid & 31;
  const int hh = lane >> 4, ln = lane & 15;

  // Load this wave's 16 Q rows as two A-fragments (k0 = 0, 32). Q pre-scaled.
  const __bf16* qptr = qb + ((size_t)bh * TSEQ + qblk + wave * 16) * HD;
  v16bf qf[2];
#pragma unroll
  for (int kf = 0; kf < 2; kf++) {
    const __bf16* p = qptr + (size_t)ln * HD + kf * 32 + hh * 8;
    qf[kf] = ldpair(p, p + 16);
  }

  v8f o[4] = {};
  float mrow[8], lrow[8];
#pragma unroll
  for (int j = 0; j < 8; j++) { mrow[j] = -__builtin_inff(); lrow[j] = 0.0f; }

  const int qrow_base = qblk + wave * 16;
  const int qlast = qblk + BQ - 1;

  for (int kb0 = 0; kb0 <= qlast; kb0 += BK) {
    __syncthreads();
    // stage K rows and V^T
    const __bf16* kg = kb + ((size_t)bh * TSEQ + kb0) * HD;
    const __bf16* vg = vb + ((size_t)bh * TSEQ + kb0) * HD;
    for (int i = tid; i < (BK * HD) / 8; i += 128) {
      *(v8bf*)(&lk[i * 8]) = *(const v8bf*)(kg + i * 8);
      v8bf t = *(const v8bf*)(vg + i * 8);
      int r = (i * 8) / HD, c = (i * 8) % HD;
#pragma unroll
      for (int e = 0; e < 8; e++) lvt[(c + e) * BK + r] = t[e];
    }
    __syncthreads();

    // S = Q * K^T : 4 column tiles x 2 k-steps
    v8f s[4];
#pragma unroll
    for (int nt = 0; nt < 4; nt++) {
      v8f a = {};
#pragma unroll
      for (int kf = 0; kf < 2; kf++) {
        const __bf16* p = &lk[(nt * 16 + ln) * HD + kf * 32 + hh * 8];
        a = wmma_bf16(qf[kf], ldpair(p, p + 16), a);
      }
      s[nt] = a;
    }

    // causal mask + online softmax (row = j + 8*hh, col = kb0 + nt*16 + ln)
#pragma unroll
    for (int j = 0; j < 8; j++) {
      int qr = qrow_base + j + 8 * hh;
      float mx = -__builtin_inff();
#pragma unroll
      for (int nt = 0; nt < 4; nt++) {
        int kc = kb0 + nt * 16 + ln;
        float sv = s[nt][j];
        if (kc > qr) sv = -__builtin_inff();
        s[nt][j] = sv;
        mx = fmaxf(mx, sv);
      }
#pragma unroll
      for (int off = 1; off < 16; off <<= 1)
        mx = fmaxf(mx, __shfl_xor(mx, off, 32));
      float newm = fmaxf(mrow[j], mx);
      float alpha = __expf(mrow[j] - newm);
      mrow[j] = newm;
      float rs = 0.0f;
#pragma unroll
      for (int nt = 0; nt < 4; nt++) {
        float p = __expf(s[nt][j] - newm);
        s[nt][j] = p;
        rs += p;
      }
#pragma unroll
      for (int off = 1; off < 16; off <<= 1) rs += __shfl_xor(rs, off, 32);
      lrow[j] = lrow[j] * alpha + rs;
#pragma unroll
      for (int dt = 0; dt < 4; dt++) o[dt][j] *= alpha;
      // stash P (bf16) for use as an A-matrix
#pragma unroll
      for (int nt = 0; nt < 4; nt++)
        lp[wave][(j + 8 * hh) * BK + nt * 16 + ln] = (__bf16)s[nt][j];
    }
    __syncthreads();

    // O += P * V  (k-dim = 64 keys, V^T rows are contiguous in n)
#pragma unroll
    for (int kf = 0; kf < 2; kf++) {
      const __bf16* pa = &lp[wave][ln * BK + kf * 32 + hh * 8];
      v16bf af = ldpair(pa, pa + 16);
#pragma unroll
      for (int dt = 0; dt < 4; dt++) {
        const __bf16* p = &lvt[(dt * 16 + ln) * BK + kf * 32 + hh * 8];
        o[dt] = wmma_bf16(af, ldpair(p, p + 16), o[dt]);
      }
    }
  }

  // normalize and write attn output in [B, T, C] (bf16) for the final GEMM
#pragma unroll
  for (int j = 0; j < 8; j++) {
    float inv = 1.0f / lrow[j];
    int t = qrow_base + j + 8 * hh;
    size_t base = ((size_t)b * TSEQ + t) * EMB + h * HD;
#pragma unroll
    for (int dt = 0; dt < 4; dt++)
      attn_out[base + dt * 16 + ln] = (__bf16)(o[dt][j] * inv);
  }
}

// ---------------------------------------------------------------------------
// host-side orchestration
// ---------------------------------------------------------------------------
extern "C" void kernel_launch(void* const* d_in, const int* in_sizes, int n_in,
                              void* d_out, int out_size, void* d_ws, size_t ws_size,
                              hipStream_t stream) {
  const float* x     = (const float*)d_in[0];   // [4, 2048, 1024]
  const float* w_qkv = (const float*)d_in[1];   // [3072, 1024]
  const float* w_out = (const float*)d_in[2];   // [1024, 1024]
  float* out = (float*)d_out;                   // [4, 2048, 1024] f32

  const int N_X  = NTOK * EMB;        // 8388608
  const int N_WQ = 3 * EMB * EMB;     // 3145728
  const int N_WO = EMB * EMB;         // 1048576
  const int N_QKV1 = BB * NH * TSEQ * HD;  // 8388608 per buffer

  char* w = (char*)d_ws;
  float* partQ  = (float*)(w + 0);
  float* partO  = (float*)(w + 1024);
  float* scaleQ = (float*)(w + 2048);
  float* scaleO = (float*)(w + 2112);
  size_t off = 4096;
  __bf16* xb   = (__bf16*)(w + off); off += (size_t)N_X * 2;
  __bf16* wqt  = (__bf16*)(w + off); off += (size_t)N_WQ * 2;
  __bf16* wot  = (__bf16*)(w + off); off += (size_t)N_WO * 2;
  __bf16* qbuf = (__bf16*)(w + off); off += (size_t)N_QKV1 * 2;
  __bf16* kbuf = (__bf16*)(w + off); off += (size_t)N_QKV1 * 2;
  __bf16* vbuf = (__bf16*)(w + off); off += (size_t)N_QKV1 * 2;
  __bf16* attb = (__bf16*)(w + off); off += (size_t)N_X * 2;

  // 1) absmean scales (deterministic)
  absum_partial<<<256, 256, 0, stream>>>(w_qkv, N_WQ, partQ);
  absum_final<<<1, 256, 0, stream>>>(partQ, N_WQ, scaleQ);
  absum_partial<<<256, 256, 0, stream>>>(w_out, N_WO, partO);
  absum_final<<<1, 256, 0, stream>>>(partO, N_WO, scaleO);

  // 2) ternary quantize + activation convert
  quantize_ternary<<<(N_WQ + 255) / 256, 256, 0, stream>>>(w_qkv, scaleQ, wqt, N_WQ);
  quantize_ternary<<<(N_WO + 255) / 256, 256, 0, stream>>>(w_out, scaleO, wot, N_WO);
  cvt_f32_bf16<<<(N_X + 255) / 256, 256, 0, stream>>>(x, xb, N_X);

  // 3) QKV projection (scatter to [B,H,T,D], Q pre-scaled by 0.125)
  gemm_qkv<<<dim3(NTOK / 128, (3 * EMB) / 128), 256, 0, stream>>>(xb, wqt, qbuf, kbuf, vbuf);

  // 4) flash attention
  flash_attn<<<dim3(BB * NH, TSEQ / BQ), 128, 0, stream>>>(qbuf, kbuf, vbuf, attb);

  // 5) output projection -> f32
  gemm_out<<<dim3(NTOK / 128, EMB / 128), 256, 0, stream>>>(attb, wot, out);
}
